// GATEncoder_12455405158966
// MI455X (gfx1250) — matrix operand order repaired and checked by reading, not compiled, not run on hardware
//
#include <hip/hip_runtime.h>

typedef __attribute__((ext_vector_type(16))) __bf16 v16bf;
typedef __attribute__((ext_vector_type(8)))  __bf16 v8bf;
typedef __attribute__((ext_vector_type(8)))  float  v8f;

#define H_  8
#define B_  32
#define N_  512
#define D_  64
#define BN  16384   // B_*N_
#define NEG_SLOPE 0.01f

static __device__ __forceinline__ __bf16 f2bf(float f) {
  unsigned u = __builtin_bit_cast(unsigned, f);
  unsigned r = u + 0x7FFFu + ((u >> 16) & 1u);   // round-to-nearest-even
  unsigned short hsh = (unsigned short)(r >> 16);
  return __builtin_bit_cast(__bf16, hsh);
}

// ---------------- elementwise f32 -> bf16 ----------------
__global__ void k_cvt(const float* __restrict__ in, __bf16* __restrict__ out, int n) {
  int i = blockIdx.x * 256 + threadIdx.x;
  if (i < n) out[i] = f2bf(in[i]);
}

// ---------------- W (H,F,D) f32 -> WT (H,D,F) bf16 (K-contiguous for B tiles)
__global__ void k_cvt_wt(const float* __restrict__ W, __bf16* __restrict__ WT, int F) {
  int i = blockIdx.x * 256 + threadIdx.x;     // over H*F*D
  if (i >= H_ * F * D_) return;
  int d = i % D_;
  int f = (i / D_) % F;
  int h = i / (D_ * F);
  WT[((size_t)h * D_ + d) * F + f] = f2bf(W[i]);
}

// ---------------- projection GEMM: fts = X @ W[h]  (per head) ----------------
// X: [BN][F] bf16 row-major (features contiguous -> A tile loads are 16B vectors)
// WT: [H][D][F] bf16 (K contiguous -> B tile loads are 32B vectors)
// fts: [H][BN][D] f32 (residual + f1/f2 source); ftsT: [H*B][D][N] bf16 (attention V)
__global__ __launch_bounds__(256) void k_proj(
    const __bf16* __restrict__ X, const __bf16* __restrict__ WT, int F,
    float* __restrict__ fts, __bf16* __restrict__ ftsT)
{
  const int h     = blockIdx.y;
  const int chunk = blockIdx.x;            // BN/128 chunks of 128 rows
  const int wave  = threadIdx.x >> 5;
  const int lane  = threadIdx.x & 31;
  const int ln    = lane & 15;
  const int hi    = lane >> 4;             // lane-group (K half selector)
  const int rbase = chunk * 128 + wave * 16;
  const __bf16* xrow = X + (size_t)(rbase + ln) * F;

  v8f acc[4] = {};
  for (int k = 0; k < F; k += 32) {
    // A tile 16x32: lane = row, 16 contiguous K starting at k + hi*16
    v8bf alo = *(const v8bf*)(xrow + k + hi * 16);
    v8bf ahi = *(const v8bf*)(xrow + k + hi * 16 + 8);
    v16bf a;
#pragma unroll
    for (int e = 0; e < 8; ++e) { a[e] = alo[e]; a[e + 8] = ahi[e]; }
#pragma unroll
    for (int t = 0; t < 4; ++t) {
      // B tile 32x16: lane = column, 16 contiguous K starting at k + hi*16
      const __bf16* wp = WT + ((size_t)h * D_ + t * 16 + ln) * F + k + hi * 16;
      v16bf bb = *(const v16bf*)wp;
      acc[t] = __builtin_amdgcn_wmma_f32_16x16x32_bf16(
          false, a, false, bb, (short)0, acc[t], false, false);
    }
  }

  const int bidx = rbase / N_;             // 128 | 512 -> whole chunk in one batch
  const int hbi  = h * B_ + bidx;
#pragma unroll
  for (int t = 0; t < 4; ++t) {
#pragma unroll
    for (int r = 0; r < 8; ++r) {
      int m    = r + hi * 8;               // C layout: VGPR r, lane-group -> row
      int grow = rbase + m;
      int dcol = t * 16 + ln;
      float v  = acc[t][r];
      fts[((size_t)h * BN + grow) * D_ + dcol] = v;
      int n = grow & (N_ - 1);
      ftsT[((size_t)hbi * D_ + dcol) * N_ + n] = f2bf(v);
    }
  }
}

// ---------------- f1/f2 = fts . a{1,2}w + bias (one wave per row) -----------
__global__ __launch_bounds__(256) void k_f12(
    const float* __restrict__ fts,
    const float* __restrict__ a1w, const float* __restrict__ a1b,
    const float* __restrict__ a2w, const float* __restrict__ a2b,
    float* __restrict__ f1, float* __restrict__ f2)
{
  int wave = threadIdx.x >> 5, lane = threadIdx.x & 31;
  int rg = blockIdx.x * 8 + wave;          // row over H*BN
  int hh = rg >> 14;                       // / BN
  const float* p = fts + (size_t)rg * D_;
  float v0 = p[lane], v1 = p[lane + 32];
  float s1 = v0 * a1w[hh * D_ + lane] + v1 * a1w[hh * D_ + lane + 32];
  float s2 = v0 * a2w[hh * D_ + lane] + v1 * a2w[hh * D_ + lane + 32];
#pragma unroll
  for (int off = 16; off; off >>= 1) {
    s1 += __shfl_xor(s1, off, 32);
    s2 += __shfl_xor(s2, off, 32);
  }
  if (lane == 0) { f1[rg] = s1 + a1b[hh]; f2[rg] = s2 + a2b[hh]; }
}

// ---------------- fused attention (flash-style, single pass) -----------------
// logits(i,j) = leaky(f1[i]+f2[j]); row max = leaky(f1[i]+max_j f2[j]) (monotone)
// out = elu( softmax(logits) @ V + fts )
template <bool FINAL>
__global__ __launch_bounds__(256) void k_attn(
    const float* __restrict__ f1g, const float* __restrict__ f2g,
    const __bf16* __restrict__ ftsT,   // [H*B][D][N] bf16 (V, K-contiguous)
    const float*  __restrict__ fts,    // [H*B][N][D] f32 (residual)
    float* __restrict__ outf, __bf16* __restrict__ outb)
{
  __shared__ __bf16 VT[D_ * N_];       // 64 KB (320 KB/WGP available)
  __shared__ float f1s[N_], f2s[N_];
  __shared__ float sMax;
  const int hbi  = blockIdx.x;
  const int tid  = threadIdx.x;
  const int wave = tid >> 5, lane = tid & 31, ln = lane & 15, hi = lane >> 4;

  { // stage V into LDS via async copy (64 KB, bypasses VGPRs, ASYNCcnt-tracked)
    const __bf16* src = ftsT + (size_t)hbi * D_ * N_;
#pragma unroll
    for (int it = 0; it < 16; ++it) {
      unsigned long long gaddr =
          (unsigned long long)(const void*)(src + (size_t)(tid + it * 256) * 8);
      unsigned ldsAddr =
          (unsigned)(unsigned long long)(const void*)(VT + (size_t)(tid + it * 256) * 8);
      asm volatile("global_load_async_to_lds_b128 %0, %1, off"
                   :: "v"(ldsAddr), "v"(gaddr) : "memory");
    }
  }
  f1s[tid]       = f1g[hbi * N_ + tid];
  f1s[tid + 256] = f1g[hbi * N_ + tid + 256];
  f2s[tid]       = f2g[hbi * N_ + tid];
  f2s[tid + 256] = f2g[hbi * N_ + tid + 256];
  asm volatile("s_wait_asynccnt 0" ::: "memory");
  __syncthreads();
  if (wave == 0) {                      // max_j f2[j]
    float m = -3.0e38f;
    for (int idx = lane; idx < N_; idx += 32) m = fmaxf(m, f2s[idx]);
#pragma unroll
    for (int off = 16; off; off >>= 1) m = fmaxf(m, __shfl_xor(m, off, 32));
    if (lane == 0) sMax = m;
  }
  __syncthreads();
  const float f2max = sMax;
  const int h = hbi >> 5, bidx = hbi & 31;
  const int koff = hi * 16;

  for (int rt = wave; rt < N_ / 16; rt += 8) {
    const int ibase = rt * 16;
    const float f1i = f1s[ibase + ln];          // A layout: lane -> row ibase+ln
    float t0 = f1i + f2max;
    const float rowmax = (t0 > 0.f) ? t0 : NEG_SLOPE * t0;
    float psum = 0.f;
    v8f acc[4] = {};
    for (int jb = 0; jb < N_; jb += 32) {
      v16bf a;
#pragma unroll
      for (int e = 0; e < 16; ++e) {            // K = e + (lane>=16?16:0)
        int jj = jb + e + koff;
        float lg = f1i + f2s[jj];
        lg = (lg > 0.f) ? lg : NEG_SLOPE * lg;
        float pexp = __expf(lg - rowmax);
        psum += pexp;
        a[e] = f2bf(pexp);
      }
#pragma unroll
      for (int t = 0; t < 4; ++t) {
        const v16bf bmat = *(const v16bf*)&VT[(t * 16 + ln) * N_ + jb + hi * 16];
        acc[t] = __builtin_amdgcn_wmma_f32_16x16x32_bf16(
            false, a, false, bmat, (short)0, acc[t], false, false);
      }
    }
    // rowsum: lane l and l+16 cover the two K halves of row ibase + l%16
    float rsAll = psum + __shfl_xor(psum, 16, 32);
#pragma unroll
    for (int t = 0; t < 4; ++t) {
#pragma unroll
      for (int r = 0; r < 8; ++r) {
        int m = r + hi * 8;
        float rs = __shfl(rsAll, m, 32);        // row m's sum lives in lane m
        int i = ibase + m;
        int dcol = t * 16 + ln;
        float v = acc[t][r] / rs + fts[((size_t)hbi * N_ + i) * D_ + dcol];
        v = (v > 0.f) ? v : (__expf(v) - 1.f);  // ELU
        size_t oi = ((size_t)(bidx * N_ + i)) * 512 + h * D_ + dcol;  // concat heads
        if (FINAL) outf[oi] = v; else outb[oi] = f2bf(v);
      }
    }
  }
}

extern "C" void kernel_launch(void* const* d_in, const int* in_sizes, int n_in,
                              void* d_out, int out_size, void* d_ws, size_t ws_size,
                              hipStream_t stream) {
  const float* x   = (const float*)d_in[0];
  const float* W0  = (const float*)d_in[1];
  const float* W1  = (const float*)d_in[2];
  const float* a1w = (const float*)d_in[3];   // (2,H,64)
  const float* a1b = (const float*)d_in[4];   // (2,H)
  const float* a2w = (const float*)d_in[5];
  const float* a2b = (const float*)d_in[6];
  float* out = (float*)d_out;

  char* ws = (char*)d_ws;
  size_t off = 0;
  auto alloc = [&](size_t bytes) {
    void* p = ws + off;
    off = (off + bytes + 255) & ~(size_t)255;
    return p;
  };
  __bf16* x_bf  = (__bf16*)alloc((size_t)BN * 128 * 2);
  __bf16* WT0   = (__bf16*)alloc((size_t)H_ * 128 * D_ * 2);
  __bf16* WT1   = (__bf16*)alloc((size_t)H_ * 512 * D_ * 2);
  float*  fts   = (float*) alloc((size_t)H_ * BN * D_ * 4);
  __bf16* ftsT  = (__bf16*)alloc((size_t)H_ * B_ * D_ * N_ * 2);
  float*  f1    = (float*) alloc((size_t)H_ * BN * 4);
  float*  f2    = (float*) alloc((size_t)H_ * BN * 4);
  __bf16* hbuf  = (__bf16*)alloc((size_t)BN * 512 * 2);
  (void)ws_size; (void)in_sizes; (void)n_in; (void)out_size;

  // convert inputs
  k_cvt<<<(BN * 128 + 255) / 256, 256, 0, stream>>>(x, x_bf, BN * 128);
  k_cvt_wt<<<(H_ * 128 * D_ + 255) / 256, 256, 0, stream>>>(W0, WT0, 128);
  k_cvt_wt<<<(H_ * 512 * D_ + 255) / 256, 256, 0, stream>>>(W1, WT1, 512);

  // ---- stack 0 ----
  k_proj<<<dim3(BN / 128, H_), 256, 0, stream>>>(x_bf, WT0, 128, fts, ftsT);
  k_f12<<<(H_ * BN) / 8, 256, 0, stream>>>(fts, a1w, a1b, a2w, a2b, f1, f2);
  k_attn<false><<<H_ * B_, 256, 0, stream>>>(f1, f2, ftsT, fts, nullptr, hbuf);

  // ---- stack 1 ----
  k_proj<<<dim3(BN / 128, H_), 256, 0, stream>>>(hbuf, WT1, 512, fts, ftsT);
  k_f12<<<(H_ * BN) / 8, 256, 0, stream>>>(fts, a1w + H_ * D_, a1b + H_,
                                           a2w + H_ * D_, a2b + H_, f1, f2);
  k_attn<true><<<H_ * B_, 256, 0, stream>>>(f1, f2, ftsT, fts, out, nullptr);
}
